// Sequence_34754875359849
// MI455X (gfx1250) — matrix operand order repaired and checked by reading, not compiled, not run on hardware
//
#include <hip/hip_runtime.h>
#include <hip/hip_fp16.h>

typedef __attribute__((ext_vector_type(16))) _Float16 v16h;
typedef __attribute__((ext_vector_type(8)))  float    v8f;

#define HID 51
#define NG  204      // 4*HID
#define NT  13       // ceil(204/16) N-tiles (padded to 208)

__device__ __forceinline__ float fsigmoid(float x) {
    return __builtin_amdgcn_rcpf(1.f + __expf(-x));
}
__device__ __forceinline__ float ftanh(float x) {
    return 2.f * __builtin_amdgcn_rcpf(1.f + __expf(-2.f * x)) - 1.f;
}

__global__ void __launch_bounds__(32)
lstm2_scan_kernel(const float* __restrict__ x,
                  const float* __restrict__ Wih1, const float* __restrict__ Whh1,
                  const float* __restrict__ bih1, const float* __restrict__ bhh1,
                  const float* __restrict__ Wih2, const float* __restrict__ Whh2,
                  const float* __restrict__ bih2, const float* __restrict__ bhh2,
                  float* __restrict__ out, int T)
{
    // A-fragments of Whh1 (G^T = Whh1 * h1^T), pre-swizzled to WMMA lane order
    __shared__ _Float16 WB[NT * 2 * 32 * 16] __attribute__((aligned(32)));   // 26 KB
    __shared__ _Float16 hbuf[16 * 64]        __attribute__((aligned(32)));   // h1 f16, K-padded
    __shared__ float gbuf[208 * 16];      // G^T staging (13 KB)
    __shared__ float c1buf[16 * 52];      // c1 for layer-2 input (pad 52 avoids bank conflict)
    __shared__ float w1x[NG], bsum1[NG];  // Wih1 (outer-product vec), bih1+bhh1
    __shared__ float w2[4 * 52], w2h[4], bsum2[4], inbuf[16];

    const int lane = threadIdx.x;     // 0..31
    const int m    = lane & 15;       // batch row within tile
    const int hi   = lane >> 4;       // half-wave
    const int m0   = blockIdx.x * 16; // global batch base

    // ---------------- one-time init: swizzle weights into fragment order -------------
    for (int idx = lane; idx < NT * 2 * 32 * 16; idx += 32) {
        int t = idx >> 10;            // tile
        int s = (idx >> 9) & 1;       // K-half
        int l = (idx >> 4) & 31;      // lane the fragment element belongs to
        int q = idx & 15;             // element within v16h
        int M = l & 15;               // A-matrix row (gate index within tile)
        int k = s * 32 + ((l >> 4) ? 16 : 0) + q;  // contiguous K per half-wave (matches B)
        int n = t * 16 + M;
        float v = (n < NG && k < HID) ? Whh1[n * HID + k] : 0.f;
        WB[idx] = (_Float16)v;
    }
    for (int idx = lane; idx < 16 * 64; idx += 32) hbuf[idx] = (_Float16)0.f;
    for (int idx = lane; idx < 16 * 52; idx += 32) c1buf[idx] = 0.f;
    for (int idx = lane; idx < NG; idx += 32) {
        w1x[idx]   = Wih1[idx];
        bsum1[idx] = bih1[idx] + bhh1[idx];
    }
    for (int idx = lane; idx < 4 * 52; idx += 32) {
        int g = idx / 52, j = idx % 52;
        w2[idx] = (j < HID) ? Wih2[g * HID + j] : 0.f;
    }
    if (lane < 4) { w2h[lane] = Whh2[lane]; bsum2[lane] = bih2[lane] + bhh2[lane]; }

    float c1reg[26];
    #pragma unroll
    for (int i = 0; i < 26; ++i) c1reg[i] = 0.f;
    float h2 = 0.f, c2 = 0.f;

    __syncthreads();  // single wave: lowers to waits only

    // ---------------- one LSTM step (input scalar per batch row is in inbuf) ---------
    auto do_step = [&]() {
        float xin = inbuf[m];

        // B-operand fragments: h1^T, contiguous per lane from row-major hbuf
        v16h bf0 = *(const v16h*)&hbuf[m * 64 + hi * 16];       // K = 0..31
        v16h bf1 = *(const v16h*)&hbuf[m * 64 + 32 + hi * 16];  // K = 32..63

        // G^T = Whh1 * h1^T : 13 tiles x 2 K-halves of v_wmma_f32_16x16x32_f16
        #pragma unroll
        for (int t = 0; t < NT; ++t) {
            v16h a0 = *(const v16h*)&WB[((t * 2 + 0) * 32 + lane) * 16];
            v16h a1 = *(const v16h*)&WB[((t * 2 + 1) * 32 + lane) * 16];
            v8f acc = {};
            acc = __builtin_amdgcn_wmma_f32_16x16x32_f16(false, a0, false, bf0,
                                                         (short)0, acc, false, false);
            acc = __builtin_amdgcn_wmma_f32_16x16x32_f16(false, a1, false, bf1,
                                                         (short)0, acc, false, false);
            // C layout: VGPR r, lane l -> row = t*16 + r + 8*hi, col = m
            #pragma unroll
            for (int r = 0; r < 8; ++r)
                gbuf[(t * 16 + r + hi * 8) * 16 + m] = acc[r];
        }

        // gate nonlinearity + cell update; lane handles (m, j = 2*jj + hi)
        for (int jj = 0; jj < 26; ++jj) {
            int j = jj * 2 + hi;
            if (j < HID) {
                float gi = gbuf[(j          ) * 16 + m] + xin * w1x[j          ] + bsum1[j          ];
                float gf = gbuf[(HID   + j  ) * 16 + m] + xin * w1x[HID   + j  ] + bsum1[HID   + j  ];
                float gg = gbuf[(2*HID + j  ) * 16 + m] + xin * w1x[2*HID + j  ] + bsum1[2*HID + j  ];
                float go = gbuf[(3*HID + j  ) * 16 + m] + xin * w1x[3*HID + j  ] + bsum1[3*HID + j  ];
                float c  = fsigmoid(gf) * c1reg[jj] + fsigmoid(gi) * ftanh(gg);
                float h  = fsigmoid(go) * ftanh(c);
                c1reg[jj]        = c;
                hbuf[m * 64 + j] = (_Float16)h;    // next step's B operand
                c1buf[m * 52 + j] = c;             // layer-2 input
            }
        }

        // layer 2 (input = c1, hidden size 1): lanes 0..15, one batch row each
        if (lane < 16) {
            float g0 = bsum2[0] + h2 * w2h[0];
            float g1 = bsum2[1] + h2 * w2h[1];
            float g2 = bsum2[2] + h2 * w2h[2];
            float g3 = bsum2[3] + h2 * w2h[3];
            for (int j = 0; j < HID; ++j) {
                float cv = c1buf[m * 52 + j];
                g0 += cv * w2[0 * 52 + j];
                g1 += cv * w2[1 * 52 + j];
                g2 += cv * w2[2 * 52 + j];
                g3 += cv * w2[3 * 52 + j];
            }
            c2 = fsigmoid(g1) * c2 + fsigmoid(g0) * ftanh(g2);  // i,f,g,o order
            h2 = fsigmoid(g3) * ftanh(c2);
        }
    };

    // ---------------- time scan ------------------------------------------------------
    for (int t = 0; t < T; ++t) {
        if (lane < 16) {
            inbuf[lane] = x[(size_t)(m0 + lane) * T + t];
            if (t + 16 < T)
                __builtin_prefetch(&x[(size_t)(m0 + lane) * T + t + 16], 0, 0);
        }
        do_step();
    }

    // extra step after the loop: input = c2; output = new c2 of layer 2
    if (lane < 16) inbuf[lane] = c2;
    do_step();

    if (lane < 16) out[m0 + lane] = c2;
}

extern "C" void kernel_launch(void* const* d_in, const int* in_sizes, int n_in,
                              void* d_out, int out_size, void* d_ws, size_t ws_size,
                              hipStream_t stream) {
    const float* x    = (const float*)d_in[0];
    const float* Wih1 = (const float*)d_in[1];
    const float* Whh1 = (const float*)d_in[2];
    const float* bih1 = (const float*)d_in[3];
    const float* bhh1 = (const float*)d_in[4];
    const float* Wih2 = (const float*)d_in[5];
    const float* Whh2 = (const float*)d_in[6];
    const float* bih2 = (const float*)d_in[7];
    const float* bhh2 = (const float*)d_in[8];
    float* out = (float*)d_out;

    const int B = out_size;             // output is [B,1]
    const int T = in_sizes[0] / B;      // x is [B,T]
    const int grid = B / 16;            // one wave32 block per 16-row batch tile

    lstm2_scan_kernel<<<grid, 32, 0, stream>>>(x, Wih1, Whh1, bih1, bhh1,
                                               Wih2, Whh2, bih2, bhh2, out, T);
}
